// RVQBottleneck_44298292691483
// MI455X (gfx1250) — matrix operand order repaired and checked by reading, not compiled, not run on hardware
//
#include <hip/hip_runtime.h>
#include <hip/hip_bf16.h>

// ---------------------------------------------------------------------------
// RVQ bottleneck for MI455X (gfx1250, wave32, WMMA).
// Distances via bf16x3 fp32-emulated v_wmma_f32_16x16x32_bf16.
// B tiles streamed via CDNA5 async global->LDS DMA when available.
// ---------------------------------------------------------------------------

#define RVQ_Q     8
#define RVQ_K     4096
#define RVQ_D     128
#define RVQ_B     8
#define RVQ_N     4096
#define RVQ_ROWS  (RVQ_B * RVQ_N)        // 32768 flat (b,n) rows
#define ROWS_PER_BLOCK 64
#define STAGE_BLOCKS   (RVQ_ROWS / ROWS_PER_BLOCK)   // 512
#define NTILES         (RVQ_K / 16)                   // 256

typedef __bf16 v16bf __attribute__((ext_vector_type(16)));
typedef float  v8f   __attribute__((ext_vector_type(8)));

union Frag16 {            // one WMMA 16-bit operand: 16 bf16 per lane = 8 VGPRs
    v16bf v;
    int4  q[2];           // two 16-byte halves for LDS b128 loads
};

// ---- CDNA5 async global->LDS path (guarded; falls back to reg prefetch) ----
#if defined(__has_builtin)
#if __has_builtin(__builtin_amdgcn_global_load_async_to_lds_b128)
#define RVQ_ASYNC_LDS 1
#endif
#endif
#ifndef RVQ_ASYNC_LDS
#define RVQ_ASYNC_LDS 0
#endif

#if RVQ_ASYNC_LDS
#if __has_builtin(__builtin_amdgcn_s_wait_asynccnt)
#define RVQ_WAIT_ASYNC(n) __builtin_amdgcn_s_wait_asynccnt(n)
#else
#define RVQ_WAIT_ASYNC(n) asm volatile("s_wait_asynccnt %0" ::"i"(n) : "memory")
#endif
typedef int rvq_v4i __attribute__((vector_size(16)));
__device__ __forceinline__ void rvq_g2l_b128(const void* g, void* l) {
    // builtin takes generic pointers to 16-byte int vectors: (src, lds_dst, off, cpol)
    __builtin_amdgcn_global_load_async_to_lds_b128((rvq_v4i*)g, (rvq_v4i*)l, 0, 0);
}
#endif

// ---------------------------------------------------------------------------
// Init: residual[b*N+n][d] = x[b][d][n]; quant = 0
// ---------------------------------------------------------------------------
__global__ __launch_bounds__(256)
void rvq1250_init(const float* __restrict__ x,
                  float* __restrict__ res,
                  float* __restrict__ quant) {
    int i = blockIdx.x * 256 + threadIdx.x;       // over B*D*N
    int n  = i & (RVQ_N - 1);
    int bd = i >> 12;
    int d  = bd & (RVQ_D - 1);
    int b  = bd >> 7;
    float v = x[i];                               // coalesced read (n fastest)
    int ro  = ((b << 12) + n) * RVQ_D + d;
    res[ro]   = v;
    quant[ro] = 0.0f;
}

// ---------------------------------------------------------------------------
// Codebook f32 -> bf16 hi/lo split (once per call), plus ||c||^2
// ---------------------------------------------------------------------------
__global__ __launch_bounds__(256)
void rvq1250_cvt(const float* __restrict__ cbs,
                 __bf16* __restrict__ cbh,
                 __bf16* __restrict__ cbl) {
    int i = blockIdx.x * 256 + threadIdx.x;       // over Q*K*D
    float v  = cbs[i];
    __bf16 h = (__bf16)v;
    cbh[i] = h;
    cbl[i] = (__bf16)(v - (float)h);
}

__global__ __launch_bounds__(256)
void rvq1250_csq(const float* __restrict__ cbs, float* __restrict__ csq) {
    int e = blockIdx.x * 256 + threadIdx.x;       // 0 .. Q*K-1
    if (e >= RVQ_Q * RVQ_K) return;
    const float* p = cbs + (size_t)e * RVQ_D;
    float s = 0.0f;
    #pragma unroll 8
    for (int d = 0; d < RVQ_D; ++d) s += p[d] * p[d];
    csq[e] = s;
}

// ---------------------------------------------------------------------------
// One RVQ stage: fused GEMM + argmin + gather + residual update.
// Block: 128 threads = 4 wave32, owns 64 rows; each wave owns a 16-row tile.
// ---------------------------------------------------------------------------
__global__ __launch_bounds__(128)
void rvq1250_stage(const float*  __restrict__ cb,    // [K][D] f32 stage codebook
                   const __bf16* __restrict__ cbh,   // [K][D] bf16 hi
                   const __bf16* __restrict__ cbl,   // [K][D] bf16 lo
                   const float*  __restrict__ csq,   // [K]    stage ||c||^2
                   float* __restrict__ res,          // [ROWS][D]
                   float* __restrict__ quant,        // [ROWS][D]
                   int*   __restrict__ idx_out,      // [ROWS][Q] (int32 bits)
                   float* __restrict__ loss_part,    // [Q][STAGE_BLOCKS]
                   int q) {
    __shared__ __attribute__((aligned(16))) __bf16 sAhi[ROWS_PER_BLOCK][RVQ_D];
    __shared__ __attribute__((aligned(16))) __bf16 sAlo[ROWS_PER_BLOCK][RVQ_D];
    __shared__ __attribute__((aligned(16))) __bf16 sBh[2][16][RVQ_D];
    __shared__ __attribute__((aligned(16))) __bf16 sBl[2][16][RVQ_D];
    __shared__ float sCsq[RVQ_K];
    __shared__ int   sIdx[ROWS_PER_BLOCK];
    __shared__ float sRed[128];

    const int tid     = threadIdx.x;
    const int rowBase = blockIdx.x * ROWS_PER_BLOCK;

    // ---- phase 0: residual rows -> hi/lo bf16 split in LDS; csq -> LDS -----
    for (int i = tid; i < ROWS_PER_BLOCK * RVQ_D; i += 128) {
        int r = i >> 7, d = i & (RVQ_D - 1);
        float v  = res[(rowBase + r) * RVQ_D + d];
        __bf16 h = (__bf16)v;
        sAhi[r][d] = h;
        sAlo[r][d] = (__bf16)(v - (float)h);
    }
    for (int i = tid; i < RVQ_K; i += 128) sCsq[i] = csq[i];
    __syncthreads();

    // ---- build per-wave A fragments (invariant over the K loop) ------------
    const int lane = tid & 31;
    const int wave = tid >> 5;
    const int half = lane >> 4;      // lane 0-15 vs 16-31
    const int ml   = lane & 15;
    const int aRow = wave * 16 + ml; // A: lane carries matrix row M

    Frag16 ahi[4], alo[4];
    #pragma unroll
    for (int c = 0; c < 4; ++c) {
        // ISA A layout (16-bit 16x32): elems 0-7 -> K = half*8 + 0..7,
        //                              elems 8-15 -> K = 16 + half*8 + 0..7
        int d0 = c * 32 + half * 8;
        ahi[c].q[0] = *(const int4*)&sAhi[aRow][d0];
        ahi[c].q[1] = *(const int4*)&sAhi[aRow][d0 + 16];
        alo[c].q[0] = *(const int4*)&sAlo[aRow][d0];
        alo[c].q[1] = *(const int4*)&sAlo[aRow][d0 + 16];
    }

    float best[8];
    int   bidx[8];
    #pragma unroll
    for (int r = 0; r < 8; ++r) { best[r] = 3.0e38f; bidx[r] = 0; }

    // B tile staging: 256 16-byte chunks per array per tile; thread copies 2+2
    const int4* gH = (const int4*)cbh;     // 16 int4 per 256-byte entry row
    const int4* gL = (const int4*)cbl;

#if RVQ_ASYNC_LDS
    // ---- async DMA double-buffered pipeline --------------------------------
    {
        int4* lh = (int4*)&sBh[0][0][0];
        int4* ll = (int4*)&sBl[0][0][0];
        rvq_g2l_b128(gH + tid,       lh + tid);
        rvq_g2l_b128(gH + tid + 128, lh + tid + 128);
        rvq_g2l_b128(gL + tid,       ll + tid);
        rvq_g2l_b128(gL + tid + 128, ll + tid + 128);
    }
#else
    int4 rh0 = gH[tid], rh1 = gH[tid + 128];
    int4 rl0 = gL[tid], rl1 = gL[tid + 128];
#endif

    for (int t = 0; t < NTILES; ++t) {
#if RVQ_ASYNC_LDS
        const int cur = t & 1;
        if (t + 1 < NTILES) {       // prefetch next tile into other buffer
            int4* lh = (int4*)&sBh[cur ^ 1][0][0];
            int4* ll = (int4*)&sBl[cur ^ 1][0][0];
            const int4* nh = gH + (t + 1) * 256;
            const int4* nl = gL + (t + 1) * 256;
            rvq_g2l_b128(nh + tid,       lh + tid);
            rvq_g2l_b128(nh + tid + 128, lh + tid + 128);
            rvq_g2l_b128(nl + tid,       ll + tid);
            rvq_g2l_b128(nl + tid + 128, ll + tid + 128);
            RVQ_WAIT_ASYNC(4);      // tile t resident; tile t+1 stays in flight
        } else {
            RVQ_WAIT_ASYNC(0);
        }
        __syncthreads();
#else
        const int cur = 0;
        __syncthreads();            // readers of sB (iter t-1) done
        {
            int4* lh = (int4*)&sBh[0][0][0];
            int4* ll = (int4*)&sBl[0][0][0];
            lh[tid] = rh0; lh[tid + 128] = rh1;
            ll[tid] = rl0; ll[tid + 128] = rl1;
        }
        if (t + 1 < NTILES) {       // global latency overlaps this tile's WMMAs
            const int4* nh = gH + (t + 1) * 256;
            const int4* nl = gL + (t + 1) * 256;
            rh0 = nh[tid]; rh1 = nh[tid + 128];
            rl0 = nl[tid]; rl1 = nl[tid + 128];
        }
        __syncthreads();
#endif

        v8f acc = {0.f, 0.f, 0.f, 0.f, 0.f, 0.f, 0.f, 0.f};
        #pragma unroll
        for (int c = 0; c < 4; ++c) {
            // ISA B layout (16-bit 32x16): lane carries column N = ml,
            // elems 0-7 -> K = half*8 + 0..7, elems 8-15 -> K = 16 + half*8 + 0..7
            Frag16 bhi, blo;
            int d0 = c * 32 + half * 8;
            bhi.q[0] = *(const int4*)&sBh[cur][ml][d0];
            bhi.q[1] = *(const int4*)&sBh[cur][ml][d0 + 16];
            blo.q[0] = *(const int4*)&sBl[cur][ml][d0];
            blo.q[1] = *(const int4*)&sBl[cur][ml][d0 + 16];
            // fp32-emulated dot: small terms first, then hi*hi
            acc = __builtin_amdgcn_wmma_f32_16x16x32_bf16(
                      false, alo[c].v, false, bhi.v, (short)0, acc, false, false);
            acc = __builtin_amdgcn_wmma_f32_16x16x32_bf16(
                      false, ahi[c].v, false, blo.v, (short)0, acc, false, false);
            acc = __builtin_amdgcn_wmma_f32_16x16x32_bf16(
                      false, ahi[c].v, false, bhi.v, (short)0, acc, false, false);
        }

        // distances: ||c||^2 - 2*dot  (||r||^2 constant per row -> dropped)
        int   entry = t * 16 + ml;          // C/D: lane carries column N
        float cs    = sCsq[entry];          // LDS hit, no VMEM stall
        #pragma unroll
        for (int r = 0; r < 8; ++r) {       // C/D: vgpr r -> M = half*8 + r
            float dist = cs - 2.0f * acc[r];
            if (dist < best[r]) { best[r] = dist; bidx[r] = entry; }
        }
#if RVQ_ASYNC_LDS
        __syncthreads();   // readers done before next iter's DMA overwrites
#endif
    }

    // ---- cross-lane argmin within each 16-lane half ------------------------
    #pragma unroll
    for (int off = 1; off < 16; off <<= 1) {
        #pragma unroll
        for (int r = 0; r < 8; ++r) {
            float ov = __shfl_xor(best[r], off, 32);
            int   oi = __shfl_xor(bidx[r], off, 32);
            if (ov < best[r] || (ov == best[r] && oi < bidx[r])) {
                best[r] = ov; bidx[r] = oi;
            }
        }
    }
    if (ml == 0) {                          // lane 0 -> rows 0..7, lane 16 -> 8..15
        #pragma unroll
        for (int r = 0; r < 8; ++r) sIdx[wave * 16 + half * 8 + r] = bidx[r];
    }
    __syncthreads();

    // ---- phase 2: gather winners, update quant/residual, loss --------------
    float lsum = 0.0f;
    const int d = tid;                      // 128 threads = one dim each
    for (int r = 0; r < ROWS_PER_BLOCK; ++r) {
        int   gi   = sIdx[r];
        float cv   = cb[gi * RVQ_D + d];
        int   grow = rowBase + r;
        float rv   = res[grow * RVQ_D + d];
        float df   = cv - rv;
        lsum      += df * df;
        quant[grow * RVQ_D + d] += cv;
        res[grow * RVQ_D + d]    = rv - cv;
    }
    if (tid < ROWS_PER_BLOCK)
        idx_out[(rowBase + tid) * RVQ_Q + q] = sIdx[tid];

    sRed[tid] = lsum;
    __syncthreads();
    #pragma unroll
    for (int s = 64; s > 0; s >>= 1) {
        if (tid < s) sRed[tid] += sRed[tid + s];
        __syncthreads();
    }
    if (tid == 0) loss_part[q * STAGE_BLOCKS + blockIdx.x] = sRed[0];
}

// ---------------------------------------------------------------------------
// Finish: out[b][d][n] = quant[b*N+n][d]
// ---------------------------------------------------------------------------
__global__ __launch_bounds__(256)
void rvq1250_out(const float* __restrict__ quant, float* __restrict__ out) {
    int i = blockIdx.x * 256 + threadIdx.x;       // over B*D*N, out-coalesced
    int n  = i & (RVQ_N - 1);
    int bd = i >> 12;
    int d  = bd & (RVQ_D - 1);
    int b  = bd >> 7;
    out[i] = quant[((b << 12) + n) * RVQ_D + d];
}

// Deterministic loss reduction (fixed order, no float atomics)
__global__ __launch_bounds__(256)
void rvq1250_loss(const float* __restrict__ loss_part, float* __restrict__ loss) {
    __shared__ float s[256];
    float a = 0.0f;
    for (int i = threadIdx.x; i < RVQ_Q * STAGE_BLOCKS; i += 256)
        a += loss_part[i];
    s[threadIdx.x] = a;
    __syncthreads();
    for (int k = 128; k > 0; k >>= 1) {
        if (threadIdx.x < k) s[threadIdx.x] += s[threadIdx.x + k];
        __syncthreads();
    }
    if (threadIdx.x == 0)
        loss[0] = s[0] / ((float)RVQ_Q * (float)RVQ_ROWS * (float)RVQ_D);
}

// ---------------------------------------------------------------------------
extern "C" void kernel_launch(void* const* d_in, const int* in_sizes, int n_in,
                              void* d_out, int out_size, void* d_ws, size_t ws_size,
                              hipStream_t stream) {
    const float* x   = (const float*)d_in[0];      // [B][D][N] f32
    const float* cbs = (const float*)d_in[1];      // [Q][K][D] f32

    // workspace carve-up (~50 MB)
    float*  res   = (float*)d_ws;                  // [ROWS][D]
    float*  quant = res   + (size_t)RVQ_ROWS * RVQ_D;
    float*  csq   = quant + (size_t)RVQ_ROWS * RVQ_D;
    float*  lossp = csq   + (size_t)RVQ_Q * RVQ_K;
    __bf16* cbh   = (__bf16*)(lossp + (size_t)RVQ_Q * STAGE_BLOCKS);
    __bf16* cbl   = cbh + (size_t)RVQ_Q * RVQ_K * RVQ_D;

    // output carve-up: out f32 | indices i32 | loss f32
    float* out  = (float*)d_out;
    int*   idx  = (int*)((float*)d_out + (size_t)RVQ_B * RVQ_D * RVQ_N);
    float* loss = (float*)d_out + (size_t)RVQ_B * RVQ_D * RVQ_N
                                + (size_t)RVQ_ROWS * RVQ_Q;

    const int elems = RVQ_B * RVQ_D * RVQ_N;       // 4,194,304
    rvq1250_init<<<elems / 256, 256, 0, stream>>>(x, res, quant);
    rvq1250_cvt<<<(RVQ_Q * RVQ_K * RVQ_D) / 256, 256, 0, stream>>>(cbs, cbh, cbl);
    rvq1250_csq<<<(RVQ_Q * RVQ_K) / 256, 256, 0, stream>>>(cbs, csq);

    for (int q = 0; q < RVQ_Q; ++q) {
        rvq1250_stage<<<STAGE_BLOCKS, 128, 0, stream>>>(
            cbs + (size_t)q * RVQ_K * RVQ_D,
            cbh + (size_t)q * RVQ_K * RVQ_D,
            cbl + (size_t)q * RVQ_K * RVQ_D,
            csq + (size_t)q * RVQ_K,
            res, quant, idx, lossp, q);
    }

    rvq1250_out<<<elems / 256, 256, 0, stream>>>(quant, out);
    rvq1250_loss<<<1, 256, 0, stream>>>(lossp, loss);
}
